// BasicBlock_45921790329685
// MI455X (gfx1250) — matrix operand order repaired and checked by reading, not compiled, not run on hardware
//
#include <hip/hip_runtime.h>
#include <math.h>

#define NB   8
#define BS   64
#define LAM  0.01f
#define BATCH 16
#define CHN  512
#define LEN  4096
#define HF   2049   // LEN/2 + 1

typedef __attribute__((ext_vector_type(16))) _Float16 v16h;
typedef __attribute__((ext_vector_type(8)))  float    v8f;

__device__ __forceinline__ v8f wmma16(v16h a, v16h b, v8f c) {
  // D = A(16x32 f16) * B(32x16 f16) + C(16x16 f32)
  return __builtin_amdgcn_wmma_f32_16x16x32_f16(false, a, false, b, (short)0, c,
                                                false, false);
}

__device__ __forceinline__ float gelu_f(float v) {
  return 0.5f * v * (1.0f + erff(v * 0.70710678118654752f));
}
__device__ __forceinline__ float sshrink(float v) {
  return v > LAM ? v - LAM : (v < -LAM ? v + LAM : 0.0f);
}

// ---------------------------------------------------------------------------
// Kernel 1: forward rfft(4096, norm='ortho') per (b,c) row.
// Stockham radix-2 autosort, LDS ping-pong (2 x 32KB). 12 stages -> natural
// order result lands back in bufA. Writes split real/imag planes (coalesced).
// ---------------------------------------------------------------------------
__global__ __launch_bounds__(256) void fft_fwd_kernel(const float* __restrict__ x,
                                                      float* __restrict__ XR,
                                                      float* __restrict__ XI) {
  __shared__ float2 bufA[LEN];
  __shared__ float2 bufB[LEN];
  const int row = blockIdx.x;
  const int tid = threadIdx.x;

  const float* xp = x + (size_t)row * LEN;
  for (int i = tid; i < LEN; i += 256) bufA[i] = make_float2(xp[i], 0.0f);
  __syncthreads();

  float2* src = bufA;
  float2* dst = bufB;
  int n = LEN, s = 1;
  while (n > 1) {
    const int m = n >> 1;
    for (int u = tid; u < LEN / 2; u += 256) {
      const int q = u & (s - 1);
      const int p = u / s;
      const float2 a = src[q + s * p];
      const float2 c = src[q + s * (p + m)];
      const float ang = -3.14159265358979323846f * (float)p / (float)m;
      float sn, cs;
      __sincosf(ang, &sn, &cs);
      const float er = a.x - c.x, ei = a.y - c.y;
      dst[q + s * (2 * p)]     = make_float2(a.x + c.x, a.y + c.y);
      dst[q + s * (2 * p + 1)] = make_float2(er * cs - ei * sn, er * sn + ei * cs);
    }
    __syncthreads();
    float2* t = src; src = dst; dst = t;
    n = m; s <<= 1;
  }
  const float inv = 1.0f / 64.0f;  // 1/sqrt(4096) ortho
  float* xr = XR + (size_t)row * HF;
  float* xi = XI + (size_t)row * HF;
  for (int h = tid; h < HF; h += 256) {
    xr[h] = src[h].x * inv;
    xi[h] = src[h].y * inv;
  }
}

// ---------------------------------------------------------------------------
// Kernel 2: fused block-diagonal complex MLP + softshrink + spectral gating.
// One workgroup = one (block k, batch b, 64-frequency tile). 4 waves, each
// wave owns 16 rows x 64 output channels via v_wmma_f32_16x16x32_f16.
// Weights (f16) + activations + o1 restage live in LDS (~83 KB / WG).
// Output overwrites the spectrum planes in place (exclusive tile ownership).
// ---------------------------------------------------------------------------
__global__ __launch_bounds__(128) void afno_mlp_kernel(
    const float* __restrict__ w1, const float* __restrict__ b1,
    const float* __restrict__ w2, const float* __restrict__ b2,
    float* __restrict__ XR, float* __restrict__ XI) {
  __shared__ __attribute__((aligned(32))) _Float16 Wlds[4][BS * BS];  // W1r W1i W2r W2i
  __shared__ float Blds[4][BS];                                       // b1r b1i b2r b2i
  __shared__ float Xr_t[BS][BS + 1];  // [in-ch][row]
  __shared__ float Xi_t[BS][BS + 1];
  __shared__ __attribute__((aligned(32))) _Float16 O1[2][BS][BS];     // [r/i][ch][row]

  const int ht = blockIdx.x, b = blockIdx.y, k = blockIdx.z;
  const int h0 = ht * 64;
  const int tid = threadIdx.x;
  const int lane = tid & 31, wv = tid >> 5;
  const int rbase = wv * 16;

  // --- stage weights (f16) and biases in LDS ---
  for (int idx = tid; idx < BS * BS; idx += 128) {
    Wlds[0][idx] = (_Float16)w1[(size_t)k * BS * BS + idx];
    Wlds[1][idx] = (_Float16)w1[(size_t)(NB + k) * BS * BS + idx];
    Wlds[2][idx] = (_Float16)w2[(size_t)k * BS * BS + idx];
    Wlds[3][idx] = (_Float16)w2[(size_t)(NB + k) * BS * BS + idx];
  }
  if (tid < BS) {
    Blds[0][tid] = b1[k * BS + tid];
    Blds[1][tid] = b1[NB * BS + k * BS + tid];
    Blds[2][tid] = b2[k * BS + tid];
    Blds[3][tid] = b2[NB * BS + k * BS + tid];
  }
  // --- coalesced load of the 64ch x 64freq activation tile ---
  for (int idx = tid; idx < BS * BS; idx += 128) {
    const int i = idx >> 6, r = idx & 63;
    const int h = h0 + r;
    float vr = 0.0f, vi = 0.0f;
    if (h < HF) {
      const size_t g = ((size_t)b * CHN + (size_t)k * BS + i) * HF + h;
      vr = XR[g];
      vi = XI[g];
    }
    Xr_t[i][r] = vr;
    Xi_t[i][r] = vi;
  }
  __syncthreads();

  // --- build layer-1 A fragments (16x32 f16 ISA layout) ---
  const int arow = rbase + (lane & 15);
  const int khi  = (lane >> 4) << 3;  // 0 or 8
  v16h aXr[2], aXi[2], aNXi[2];
#pragma unroll
  for (int ks = 0; ks < 2; ++ks) {
#pragma unroll
    for (int e = 0; e < 16; ++e) {
      const int kk = ks * 32 + ((e >> 3) << 4) + khi + (e & 7);
      const float fr = Xr_t[kk][arow];
      const float fi = Xi_t[kk][arow];
      aXr[ks][e]  = (_Float16)fr;
      aXi[ks][e]  = (_Float16)fi;
      aNXi[ks][e] = (_Float16)(-fi);
    }
  }

  // --- layer 1: o1 = gelu(Xr*W1r - Xi*W1i + b1r , Xi*W1r + Xr*W1i + b1i) ---
  v8f accR[4], accI[4];
#pragma unroll
  for (int nt = 0; nt < 4; ++nt) {
    v8f z = {0, 0, 0, 0, 0, 0, 0, 0};
    accR[nt] = z; accI[nt] = z;
#pragma unroll
    for (int ks = 0; ks < 2; ++ks) {
      // B fragment: lane%16 = N column, 16 halves = K rows (same k order as A)
      v16h bwr, bwi;
#pragma unroll
      for (int e = 0; e < 16; ++e) {
        const int kk = ks * 32 + ((e >> 3) << 4) + khi + (e & 7);
        bwr[e] = Wlds[0][kk * BS + nt * 16 + (lane & 15)];
        bwi[e] = Wlds[1][kk * BS + nt * 16 + (lane & 15)];
      }
      accR[nt] = wmma16(aXr[ks],  bwr, accR[nt]);
      accR[nt] = wmma16(aNXi[ks], bwi, accR[nt]);
      accI[nt] = wmma16(aXi[ks],  bwr, accI[nt]);
      accI[nt] = wmma16(aXr[ks],  bwi, accI[nt]);
    }
  }
  // bias + exact GELU, restage o1 to LDS for the D->A relayout
#pragma unroll
  for (int nt = 0; nt < 4; ++nt) {
    const int o = nt * 16 + (lane & 15);
#pragma unroll
    for (int j = 0; j < 8; ++j) {
      const int M = ((lane >> 4) << 3) + j;
      O1[0][o][rbase + M] = (_Float16)gelu_f(accR[nt][j] + Blds[0][o]);
      O1[1][o][rbase + M] = (_Float16)gelu_f(accI[nt][j] + Blds[1][o]);
    }
  }
  __syncthreads();

  // --- layer-2 A fragments from o1 ---
  v16h aOr[2], aOi[2], aNOi[2];
#pragma unroll
  for (int ks = 0; ks < 2; ++ks) {
#pragma unroll
    for (int e = 0; e < 16; ++e) {
      const int kk = ks * 32 + ((e >> 3) << 4) + khi + (e & 7);
      const _Float16 fr = O1[0][kk][arow];
      const _Float16 fi = O1[1][kk][arow];
      aOr[ks][e]  = fr;
      aOi[ks][e]  = fi;
      aNOi[ks][e] = (_Float16)(-(float)fi);
    }
  }
  // --- layer 2 ---
#pragma unroll
  for (int nt = 0; nt < 4; ++nt) {
    v8f z = {0, 0, 0, 0, 0, 0, 0, 0};
    accR[nt] = z; accI[nt] = z;
#pragma unroll
    for (int ks = 0; ks < 2; ++ks) {
      v16h bwr, bwi;
#pragma unroll
      for (int e = 0; e < 16; ++e) {
        const int kk = ks * 32 + ((e >> 3) << 4) + khi + (e & 7);
        bwr[e] = Wlds[2][kk * BS + nt * 16 + (lane & 15)];
        bwi[e] = Wlds[3][kk * BS + nt * 16 + (lane & 15)];
      }
      accR[nt] = wmma16(aOr[ks],  bwr, accR[nt]);
      accR[nt] = wmma16(aNOi[ks], bwi, accR[nt]);
      accI[nt] = wmma16(aOi[ks],  bwr, accI[nt]);
      accI[nt] = wmma16(aOr[ks],  bwi, accI[nt]);
    }
  }
  // --- bias + softshrink + spectral gating; update tile in place ---
#pragma unroll
  for (int nt = 0; nt < 4; ++nt) {
    const int o = nt * 16 + (lane & 15);
#pragma unroll
    for (int j = 0; j < 8; ++j) {
      const int M = ((lane >> 4) << 3) + j;
      const int r = rbase + M;
      const float o2r = sshrink(accR[nt][j] + Blds[2][o]);
      const float o2i = sshrink(accI[nt][j] + Blds[3][o]);
      const float xr = Xr_t[o][r], xi = Xi_t[o][r];
      Xr_t[o][r] = o2r * xr - o2i * xi;
      Xi_t[o][r] = o2r * xi + o2i * xr;
    }
  }
  __syncthreads();

  // --- coalesced in-place writeback of the gated spectrum ---
  for (int idx = tid; idx < BS * BS; idx += 128) {
    const int i = idx >> 6, r = idx & 63;
    const int h = h0 + r;
    if (h < HF) {
      const size_t g = ((size_t)b * CHN + (size_t)k * BS + i) * HF + h;
      XR[g] = Xr_t[i][r];
      XI[g] = Xi_t[i][r];
    }
  }
}

// ---------------------------------------------------------------------------
// Kernel 3: irfft(4096, norm='ortho') per row + residual add.
// irfft(Y) = Re( FFT_fwd(Z) )/sqrt(N) with Z[n]=conj(Y[n]) (n<=N/2) and
// Z[n]=Y[N-n] (n>N/2). Reuses the same Stockham core.
// ---------------------------------------------------------------------------
__global__ __launch_bounds__(256) void fft_inv_kernel(const float* __restrict__ YR,
                                                      const float* __restrict__ YI,
                                                      const float* __restrict__ x0,
                                                      float* __restrict__ y) {
  __shared__ float2 bufA[LEN];
  __shared__ float2 bufB[LEN];
  const int row = blockIdx.x;
  const int tid = threadIdx.x;

  const float* yr = YR + (size_t)row * HF;
  const float* yi = YI + (size_t)row * HF;
  for (int i = tid; i < LEN; i += 256) {
    float zr, zi;
    if (i < HF) { zr = yr[i];       zi = -yi[i];      }
    else        { zr = yr[LEN - i]; zi = yi[LEN - i]; }
    bufA[i] = make_float2(zr, zi);
  }
  __syncthreads();

  float2* src = bufA;
  float2* dst = bufB;
  int n = LEN, s = 1;
  while (n > 1) {
    const int m = n >> 1;
    for (int u = tid; u < LEN / 2; u += 256) {
      const int q = u & (s - 1);
      const int p = u / s;
      const float2 a = src[q + s * p];
      const float2 c = src[q + s * (p + m)];
      const float ang = -3.14159265358979323846f * (float)p / (float)m;
      float sn, cs;
      __sincosf(ang, &sn, &cs);
      const float er = a.x - c.x, ei = a.y - c.y;
      dst[q + s * (2 * p)]     = make_float2(a.x + c.x, a.y + c.y);
      dst[q + s * (2 * p + 1)] = make_float2(er * cs - ei * sn, er * sn + ei * cs);
    }
    __syncthreads();
    float2* t = src; src = dst; dst = t;
    n = m; s <<= 1;
  }
  const float inv = 1.0f / 64.0f;
  const float* xp = x0 + (size_t)row * LEN;
  float* yp = y + (size_t)row * LEN;
  for (int i = tid; i < LEN; i += 256) yp[i] = src[i].x * inv + xp[i];
}

// ---------------------------------------------------------------------------
extern "C" void kernel_launch(void* const* d_in, const int* in_sizes, int n_in,
                              void* d_out, int out_size, void* d_ws, size_t ws_size,
                              hipStream_t stream) {
  const float* x  = (const float*)d_in[0];
  const float* w1 = (const float*)d_in[1];
  const float* b1 = (const float*)d_in[2];
  const float* w2 = (const float*)d_in[3];
  const float* b2 = (const float*)d_in[4];
  float* out = (float*)d_out;

  // workspace: real + imag spectrum planes, each B*C*HF floats (~64 MB)
  float* XR = (float*)d_ws;
  float* XI = XR + (size_t)BATCH * CHN * HF;

  fft_fwd_kernel<<<BATCH * CHN, 256, 0, stream>>>(x, XR, XI);

  dim3 grid((HF + 63) / 64, BATCH, NB);  // 33 x 16 x 8
  afno_mlp_kernel<<<grid, 128, 0, stream>>>(w1, b1, w2, b2, XR, XI);

  fft_inv_kernel<<<BATCH * CHN, 256, 0, stream>>>(XR, XI, x, out);
}
